// NaiveInterferenceSelfAttention_37203006718043
// MI455X (gfx1250) — matrix operand order repaired and checked by reading, not compiled, not run on hardware
//
#include <hip/hip_runtime.h>
#include <math.h>

// Problem constants (B,S,D,H from reference)
#define Bb 2
#define Ss 2048
#define Dd 1024
#define Hh 16
#define HDv 64
#define BS (Bb * Ss)

typedef __attribute__((ext_vector_type(16))) __bf16 v16bf;
typedef __attribute__((ext_vector_type(8)))  __bf16 v8bf;
typedef __attribute__((ext_vector_type(4)))  __bf16 v4bf;
typedef __attribute__((ext_vector_type(8)))  float  v8f;

// GCC-style vector-of-4-int to match the async builtin's parameter type
typedef int i4 __attribute__((vector_size(16)));
typedef __attribute__((address_space(1))) i4* i4_gp;   // global
typedef __attribute__((address_space(3))) i4* i4_lp;   // LDS

#if __has_builtin(__builtin_amdgcn_global_load_async_to_lds_b128) && \
    __has_builtin(__builtin_amdgcn_s_wait_asynccnt)
#define USE_ASYNC_LDS 1
#else
#define USE_ASYNC_LDS 0
#endif

static __device__ __forceinline__ v16bf cat16(v8bf lo, v8bf hi) {
  return __builtin_shufflevector(lo, hi, 0,1,2,3,4,5,6,7,8,9,10,11,12,13,14,15);
}

static __device__ __forceinline__ v8f wmma_bf16(v16bf a, v16bf b, v8f c) {
  // D = A(16x32 bf16) * B(32x16 bf16) + C(16x16 f32)
  return __builtin_amdgcn_wmma_f32_16x16x32_bf16(false, a, false, b, (short)0, c, false, false);
}

// Branch-free softplus: max(a,0) + log(1+exp(-|a|)), stable for all a.
static __device__ __forceinline__ float softplus_fast(float a) {
  return fmaxf(a, 0.f) + __logf(1.f + __expf(-fabsf(a)));
}

// Branch-free tanh: copysign(1 - 2/(exp(2|p|)+1), p); exp->inf gives exactly 1.
static __device__ __forceinline__ float tanh_fast(float p) {
  float t = __expf(2.f * fabsf(p));
  float th = 1.f - 2.f * __builtin_amdgcn_rcpf(t + 1.f);
  return copysignf(th, p);
}

#if USE_ASYNC_LDS
// One instruction moves 32 lanes x 16B, tracked by ASYNCcnt.
static __device__ __forceinline__ void async_cp16(const __bf16* g, __bf16* l) {
  __builtin_amdgcn_global_load_async_to_lds_b128(
      (i4_gp)(i4*)const_cast<__bf16*>(g),
      (i4_lp)(i4*)l,
      0, 0);
}
#endif

// ---------------------------------------------------------------------------
// fp32 -> bf16 conversion (vectorized 4-wide)
// ---------------------------------------------------------------------------
__global__ void __launch_bounds__(256) cvt_bf16_kernel(const float* __restrict__ in,
                                                       __bf16* __restrict__ out, int n4) {
  int i = blockIdx.x * 256 + threadIdx.x;
  if (i >= n4) return;
  const float4 f = ((const float4*)in)[i];
  v4bf o;
  o[0] = (__bf16)f.x; o[1] = (__bf16)f.y; o[2] = (__bf16)f.z; o[3] = (__bf16)f.w;
  *(v4bf*)(out + (size_t)i * 4) = o;
}

// ---------------------------------------------------------------------------
// Fused dual projection: amp = x@Wa^T, phi = x@Wp^T, then
// eff[...,0:64] = softplus(amp)*cos(pi*tanh(phi)), eff[...,64:128] = ...*sin(...)
// One wave computes a 64x16 output strip for BOTH GEMMs: the B (weight)
// operand load is shared by 4 row-subtiles -> 8 WMMAs per 10 vector loads.
// eff layout: [B, H, S, 128] bf16
// ---------------------------------------------------------------------------
__global__ void __launch_bounds__(256) proj_amp_phi_kernel(
    const __bf16* __restrict__ xbf,     // [BS, D]
    const __bf16* __restrict__ wamp,    // [D, D]
    const __bf16* __restrict__ wphi,    // [D, D]
    __bf16* __restrict__ eff)           // [B, H, S, 128]
{
  const int lane = threadIdx.x & 31;
  const int widx = threadIdx.x >> 5;
  const int tile = blockIdx.x * 8 + widx;     // (BS/64)*(D/16) = 64*64 tiles
  const int r0 = (tile >> 6) * 64;            // 64-row strip in BS
  const int c0 = (tile & 63) * 16;            // 16-col tile in D
  const int m = lane & 15, h = lane >> 4;

  const __bf16* wa = wamp + (size_t)(c0 + m) * Dd;   // B lane n == lane&15
  const __bf16* wp = wphi + (size_t)(c0 + m) * Dd;
  const __bf16* ar[4];
#pragma unroll
  for (int rt = 0; rt < 4; ++rt) ar[rt] = xbf + (size_t)(r0 + rt * 16 + m) * Dd;

  v8f aacc[4] = {}; v8f pacc[4] = {};
  for (int kc = 0; kc < Dd; kc += 32) {
    v16bf Ba = *(const v16bf*)(wa + kc + 16 * h);
    v16bf Bp = *(const v16bf*)(wp + kc + 16 * h);
#pragma unroll
    for (int rt = 0; rt < 4; ++rt) {
      __builtin_prefetch(ar[rt] + kc + 256, 0, 0);
      v16bf A = cat16(*(const v8bf*)(ar[rt] + kc + 16 * h),
                      *(const v8bf*)(ar[rt] + kc + 16 * h + 8));
      aacc[rt] = wmma_bf16(A, Ba, aacc[rt]);
      pacc[rt] = wmma_bf16(A, Bp, pacc[rt]);
    }
  }

#pragma unroll
  for (int rt = 0; rt < 4; ++rt) {
#pragma unroll
    for (int r = 0; r < 8; ++r) {
      const float amp = softplus_fast(aacc[rt][r]);
      const float phi = 3.14159265358979f * tanh_fast(pacc[rt][r]);
      float sn, cs;
      __sincosf(phi, &sn, &cs);
      const int row  = r0 + rt * 16 + r + 8 * h;   // flattened B*S row
      const int bidx = row >> 11;                  // S = 2048
      const int sidx = row & 2047;
      const int d = c0 + m;
      const int head = d >> 6, dh = d & 63;
      const size_t base = (((size_t)bidx * Hh + head) * Ss + sidx) * 128;
      eff[base + dh]      = (__bf16)(amp * cs);
      eff[base + 64 + dh] = (__bf16)(amp * sn);
    }
  }
}

// ---------------------------------------------------------------------------
// V projection: v = x@Wv^T, stored TRANSPOSED per head: Vt[B,H,64,S] bf16
// ---------------------------------------------------------------------------
__global__ void __launch_bounds__(256) proj_v_kernel(
    const __bf16* __restrict__ xbf,   // [BS, D]
    const __bf16* __restrict__ wv,    // [D, D]
    __bf16* __restrict__ vt)          // [B, H, 64, S]
{
  const int lane = threadIdx.x & 31;
  const int widx = threadIdx.x >> 5;
  const int tile = blockIdx.x * 8 + widx;
  const int r0 = (tile >> 6) * 64;
  const int c0 = (tile & 63) * 16;
  const int m = lane & 15, h = lane >> 4;

  const __bf16* wr = wv + (size_t)(c0 + m) * Dd;
  const __bf16* ar[4];
#pragma unroll
  for (int rt = 0; rt < 4; ++rt) ar[rt] = xbf + (size_t)(r0 + rt * 16 + m) * Dd;

  v8f acc[4] = {};
  for (int kc = 0; kc < Dd; kc += 32) {
    v16bf Bv = *(const v16bf*)(wr + kc + 16 * h);
#pragma unroll
    for (int rt = 0; rt < 4; ++rt) {
      __builtin_prefetch(ar[rt] + kc + 256, 0, 0);
      v16bf A = cat16(*(const v8bf*)(ar[rt] + kc + 16 * h),
                      *(const v8bf*)(ar[rt] + kc + 16 * h + 8));
      acc[rt] = wmma_bf16(A, Bv, acc[rt]);
    }
  }

#pragma unroll
  for (int rt = 0; rt < 4; ++rt) {
#pragma unroll
    for (int r = 0; r < 8; ++r) {
      const int row  = r0 + rt * 16 + r + 8 * h;
      const int bidx = row >> 11;
      const int sidx = row & 2047;
      const int d = c0 + m;
      const int head = d >> 6, dh = d & 63;
      vt[(((size_t)bidx * Hh + head) * HDv + dh) * Ss + sidx] = (__bf16)acc[rt][r];
    }
  }
}

// ---------------------------------------------------------------------------
// Flash attention (causal). 8 waves of a block cooperate on 128 consecutive
// queries of one (b,h); K (32x128) and V (64x32) tiles are staged into LDS,
// double-buffered with CDNA5 async-to-LDS copies when available.
// head_dim_qk = 128 (Qeff/Keff), head_dim_v = 64.
// aout layout: merged heads [B*S, D] bf16
// ---------------------------------------------------------------------------
__global__ void __launch_bounds__(256) attn_kernel(
    const __bf16* __restrict__ Qe,    // [B*H, S, 128]
    const __bf16* __restrict__ Ke,    // [B*H, S, 128]
    const __bf16* __restrict__ Vt,    // [B*H, 64, S]
    __bf16* __restrict__ aout)        // [B*S, D]
{
  __shared__ __align__(64) __bf16 kbuf[2][32][128];  // 16 KB, keys x d_qk
  __shared__ __align__(64) __bf16 vbuf[2][64][32];   //  8 KB, d_v  x keys
  __shared__ __align__(64) __bf16 pbuf[8][16][32];   //  8 KB, per-wave P tile

  const int tid  = threadIdx.x;
  const int lane = tid & 31;
  const int widx = tid >> 5;
  const int blk  = blockIdx.x;            // B*H*(S/128) = 512 blocks
  const int bh   = blk >> 4;              // 16 q-groups per (b,h)
  const int qbase = (blk & 15) * 128;
  const int q0   = qbase + widx * 16;     // this wave's 16-query tile
  const int b    = bh >> 4;               // H = 16
  const int head = bh & 15;
  const int m = lane & 15, h = lane >> 4;

  const __bf16* Qbh = Qe + (size_t)bh * Ss * 128;
  const __bf16* Kbh = Ke + (size_t)bh * Ss * 128;
  const __bf16* Vbh = Vt + (size_t)bh * HDv * Ss;

  // Preload Q A-operands for the whole key loop (4 K-chunks of 32 over d=128)
  v16bf Aq[4];
  {
    const __bf16* qrow = Qbh + (size_t)(q0 + m) * 128;
#pragma unroll
    for (int c = 0; c < 4; ++c) {
      v8bf lo = *(const v8bf*)(qrow + 32 * c + 16 * h);
      v8bf hi = *(const v8bf*)(qrow + 32 * c + 16 * h + 8);
      Aq[c] = cat16(lo, hi);
    }
  }

  float mrow[8], lrow[8];
  v8f O[4] = {};
#pragma unroll
  for (int r = 0; r < 8; ++r) { mrow[r] = -1e30f; lrow[r] = 0.f; }

  // Block-cooperative K/V staging: K = 512 x 16B segments (2/thread),
  // V = 256 x 16B segments (1/thread).
#if USE_ASYNC_LDS
  auto stage = [&](int k0s, int bp) {
    int seg = tid;
    async_cp16(Kbh + (size_t)(k0s + (seg >> 4)) * 128 + (seg & 15) * 8,
               &kbuf[bp][seg >> 4][(seg & 15) * 8]);
    seg = tid + 256;
    async_cp16(Kbh + (size_t)(k0s + (seg >> 4)) * 128 + (seg & 15) * 8,
               &kbuf[bp][seg >> 4][(seg & 15) * 8]);
    async_cp16(Vbh + (size_t)(tid >> 2) * Ss + k0s + (tid & 3) * 8,
               &vbuf[bp][tid >> 2][(tid & 3) * 8]);
  };
#else
  auto stage = [&](int k0s, int bp) {
    int seg = tid;
    *(v8bf*)&kbuf[bp][seg >> 4][(seg & 15) * 8] =
        *(const v8bf*)(Kbh + (size_t)(k0s + (seg >> 4)) * 128 + (seg & 15) * 8);
    seg = tid + 256;
    *(v8bf*)&kbuf[bp][seg >> 4][(seg & 15) * 8] =
        *(const v8bf*)(Kbh + (size_t)(k0s + (seg >> 4)) * 128 + (seg & 15) * 8);
    *(v8bf*)&vbuf[bp][tid >> 2][(tid & 3) * 8] =
        *(const v8bf*)(Vbh + (size_t)(tid >> 2) * Ss + k0s + (tid & 3) * 8);
  };
#endif

  const int ntrips = (qbase + 128) >> 5;   // 32-key blocks, causal bound
#if USE_ASYNC_LDS
  stage(0, 0);                             // prologue: block 0 -> buffer 0
#endif

  for (int it = 0; it < ntrips; ++it) {
    const int k0  = it << 5;
    const int cur = it & 1;
#if USE_ASYNC_LDS
    __builtin_amdgcn_s_wait_asynccnt(0);   // my copies for buffer `cur` done
    __syncthreads();                       // everyone's copies visible
    if (it + 1 < ntrips) stage((it + 1) << 5, cur ^ 1);
#else
    __syncthreads();
    stage(k0, cur);
    __syncthreads();
#endif

    if (k0 < q0 + 16) {                    // causal: this wave needs the block
      // S tile: 16 queries x 32 keys (two 16x16 D tiles, 8 WMMAs over d=128)
      v8f s0 = {}; v8f s1 = {};
      const __bf16* kb0 = &kbuf[cur][m][0];
      const __bf16* kb1 = &kbuf[cur][16 + m][0];
#pragma unroll
      for (int c = 0; c < 4; ++c) {
        v16bf B0 = *(const v16bf*)(kb0 + 32 * c + 16 * h);
        v16bf B1 = *(const v16bf*)(kb1 + 32 * c + 16 * h);
        s0 = wmma_bf16(Aq[c], B0, s0);
        s1 = wmma_bf16(Aq[c], B1, s1);
      }

      // Online softmax per query row (row = r + 8h, cols striped over lanes)
#pragma unroll
      for (int r = 0; r < 8; ++r) {
        const int q = q0 + r + 8 * h;
        float x0 = s0[r] * 0.125f;              // 1/sqrt(64)
        float x1 = s1[r] * 0.125f;
        if (k0 + m > q)      x0 = -1e30f;       // causal mask
        if (k0 + 16 + m > q) x1 = -1e30f;
        float mx = fmaxf(x0, x1);
#pragma unroll
        for (int off = 8; off; off >>= 1)
          mx = fmaxf(mx, __shfl_xor(mx, off, 32));
        const float mnew = fmaxf(mrow[r], mx);
        const float corr = __expf(mrow[r] - mnew);
        const float p0 = __expf(x0 - mnew);
        const float p1 = __expf(x1 - mnew);
        float ps = p0 + p1;
#pragma unroll
        for (int off = 8; off; off >>= 1)
          ps += __shfl_xor(ps, off, 32);
        lrow[r] = lrow[r] * corr + ps;
        mrow[r] = mnew;
#pragma unroll
        for (int t = 0; t < 4; ++t) O[t][r] *= corr;
        pbuf[widx][r + 8 * h][m]      = (__bf16)p0;   // re-layout P via LDS
        pbuf[widx][r + 8 * h][16 + m] = (__bf16)p1;
      }
      __builtin_amdgcn_wave_barrier();

      // A-operand for P (16x32) from LDS, then O += P * V_block (4 WMMAs)
      v16bf Ap = cat16(*(const v8bf*)&pbuf[widx][m][16 * h],
                       *(const v8bf*)&pbuf[widx][m][16 * h + 8]);
#pragma unroll
      for (int t = 0; t < 4; ++t) {
        v16bf Bv = *(const v16bf*)&vbuf[cur][t * 16 + m][16 * h];
        O[t] = wmma_bf16(Ap, Bv, O[t]);
      }
      __builtin_amdgcn_wave_barrier();
    }
  }

  // Normalize and write merged-head output (bf16 for the final WMMA GEMM)
#pragma unroll
  for (int r = 0; r < 8; ++r) {
    const float inv = __builtin_amdgcn_rcpf(lrow[r]);
    const int q = q0 + r + 8 * h;
    const size_t orow = ((size_t)b * Ss + q) * Dd + head * HDv;
#pragma unroll
    for (int t = 0; t < 4; ++t)
      aout[orow + t * 16 + m] = (__bf16)(O[t][r] * inv);
  }
}

// ---------------------------------------------------------------------------
// Output projection: out = aout @ Wo^T, fp32 result (64x16 strip per wave)
// ---------------------------------------------------------------------------
__global__ void __launch_bounds__(256) out_proj_kernel(
    const __bf16* __restrict__ aout,  // [BS, D]
    const __bf16* __restrict__ wo,    // [D, D]
    float* __restrict__ out)          // [BS, D] fp32
{
  const int lane = threadIdx.x & 31;
  const int widx = threadIdx.x >> 5;
  const int tile = blockIdx.x * 8 + widx;
  const int r0 = (tile >> 6) * 64;
  const int c0 = (tile & 63) * 16;
  const int m = lane & 15, h = lane >> 4;

  const __bf16* wr = wo + (size_t)(c0 + m) * Dd;
  const __bf16* ar[4];
#pragma unroll
  for (int rt = 0; rt < 4; ++rt) ar[rt] = aout + (size_t)(r0 + rt * 16 + m) * Dd;

  v8f acc[4] = {};
  for (int kc = 0; kc < Dd; kc += 32) {
    v16bf Bw = *(const v16bf*)(wr + kc + 16 * h);
#pragma unroll
    for (int rt = 0; rt < 4; ++rt) {
      __builtin_prefetch(ar[rt] + kc + 256, 0, 0);
      v16bf A = cat16(*(const v8bf*)(ar[rt] + kc + 16 * h),
                      *(const v8bf*)(ar[rt] + kc + 16 * h + 8));
      acc[rt] = wmma_bf16(A, Bw, acc[rt]);
    }
  }

#pragma unroll
  for (int rt = 0; rt < 4; ++rt)
#pragma unroll
    for (int r = 0; r < 8; ++r)
      out[(size_t)(r0 + rt * 16 + r + 8 * h) * Dd + c0 + m] = acc[rt][r];
}

// ---------------------------------------------------------------------------
// Workspace layout (bytes); total 68 MB
// ---------------------------------------------------------------------------
static constexpr size_t OFF_XBF = 0;                      // 8 MB  [BS,D] bf16
static constexpr size_t OFF_WQA = (size_t)8  << 20;       // 2 MB each weight
static constexpr size_t OFF_WKA = (size_t)10 << 20;
static constexpr size_t OFF_WQP = (size_t)12 << 20;
static constexpr size_t OFF_WKP = (size_t)14 << 20;
static constexpr size_t OFF_WV  = (size_t)16 << 20;
static constexpr size_t OFF_WO  = (size_t)18 << 20;
static constexpr size_t OFF_QE  = (size_t)20 << 20;       // 16 MB [B,H,S,128]
static constexpr size_t OFF_KE  = (size_t)36 << 20;       // 16 MB
static constexpr size_t OFF_VT  = (size_t)52 << 20;       // 8 MB  [B,H,64,S]
static constexpr size_t OFF_AO  = (size_t)60 << 20;       // 8 MB  [BS,D] bf16

extern "C" void kernel_launch(void* const* d_in, const int* in_sizes, int n_in,
                              void* d_out, int out_size, void* d_ws, size_t ws_size,
                              hipStream_t stream) {
  (void)in_sizes; (void)n_in; (void)out_size; (void)ws_size;
  const float* x   = (const float*)d_in[0];
  const float* wqa = (const float*)d_in[1];
  const float* wka = (const float*)d_in[2];
  const float* wqp = (const float*)d_in[3];
  const float* wkp = (const float*)d_in[4];
  const float* wv  = (const float*)d_in[5];
  const float* wo  = (const float*)d_in[6];
  float* out = (float*)d_out;
  char* ws = (char*)d_ws;

  __bf16* xbf  = (__bf16*)(ws + OFF_XBF);
  __bf16* wqab = (__bf16*)(ws + OFF_WQA);
  __bf16* wkab = (__bf16*)(ws + OFF_WKA);
  __bf16* wqpb = (__bf16*)(ws + OFF_WQP);
  __bf16* wkpb = (__bf16*)(ws + OFF_WKP);
  __bf16* wvb  = (__bf16*)(ws + OFF_WV);
  __bf16* wob  = (__bf16*)(ws + OFF_WO);
  __bf16* qe   = (__bf16*)(ws + OFF_QE);
  __bf16* ke   = (__bf16*)(ws + OFF_KE);
  __bf16* vt   = (__bf16*)(ws + OFF_VT);
  __bf16* ao   = (__bf16*)(ws + OFF_AO);

  // 1) fp32 -> bf16 conversions
  const int xn4 = BS * Dd / 4;
  cvt_bf16_kernel<<<(xn4 + 255) / 256, 256, 0, stream>>>(x, xbf, xn4);
  const int wn4 = Dd * Dd / 4;
  const int wblk = (wn4 + 255) / 256;
  cvt_bf16_kernel<<<wblk, 256, 0, stream>>>(wqa, wqab, wn4);
  cvt_bf16_kernel<<<wblk, 256, 0, stream>>>(wka, wkab, wn4);
  cvt_bf16_kernel<<<wblk, 256, 0, stream>>>(wqp, wqpb, wn4);
  cvt_bf16_kernel<<<wblk, 256, 0, stream>>>(wkp, wkpb, wn4);
  cvt_bf16_kernel<<<wblk, 256, 0, stream>>>(wv,  wvb,  wn4);
  cvt_bf16_kernel<<<wblk, 256, 0, stream>>>(wo,  wob,  wn4);

  // 2) projections (4096 wave-tiles of 64x16, 8 waves/block -> 512 blocks)
  proj_amp_phi_kernel<<<512, 256, 0, stream>>>(xbf, wqab, wqpb, qe);
  proj_amp_phi_kernel<<<512, 256, 0, stream>>>(xbf, wkab, wkpb, ke);
  proj_v_kernel<<<512, 256, 0, stream>>>(xbf, wvb, vt);

  // 3) fused causal flash attention (512 blocks x 128 queries)
  attn_kernel<<<512, 256, 0, stream>>>(qe, ke, vt, ao);

  // 4) output projection
  out_proj_kernel<<<512, 256, 0, stream>>>(ao, wob, out);
}
